// new_pass_FP_GAT_later_layer_88364657148418
// MI455X (gfx1250) — compile-verified, matrix operand order checked by reading, not run on hardware
//
#include <hip/hip_runtime.h>

// ---------------------------------------------------------------------------
// GAT layer for MI455X (gfx1250, wave32). All large GEMMs run on
// v_wmma_f32_16x16x32_bf16 with fp32 accumulation. Workspace ~136 MB.
// ---------------------------------------------------------------------------

typedef __bf16 bf16_t;
typedef bf16_t v16bf __attribute__((ext_vector_type(16)));
typedef float  v8f   __attribute__((ext_vector_type(8)));

#define B_ 128
#define N_ 256
#define D_ 256
#define H_ 4
#define NEG_BIG_F (-1.0e10f)
#define LROW 272   // LDS row stride in bf16 elems: 272*2 = 544 B = 17*32 (32B aligned)

static __device__ __forceinline__ v8f wmma_bf16(v16bf a, v16bf b, v8f c) {
  // D = A(16x32) * B(32x16) + C, fp32 accumulate
  return __builtin_amdgcn_wmma_f32_16x16x32_bf16(false, a, false, b, (short)0, c,
                                                 false, false);
}

// ---------------------------------------------------------------------------
// Prep kernels: bf16 copies / transposes + diagonal extraction
// ---------------------------------------------------------------------------
__global__ void k_prep_xbf(const float* __restrict__ x, bf16_t* __restrict__ xbf,
                           float* __restrict__ diag) {
  size_t i = (size_t)blockIdx.x * blockDim.x + threadIdx.x;
  if (i < (size_t)B_ * N_ * N_) xbf[i] = (bf16_t)x[i];   // 0/1 exact in bf16
  if (i < (size_t)B_ * N_) {
    size_t b = i / N_, n = i % N_;
    diag[i] = x[(b * N_ + n) * N_ + n];
  }
}

__global__ void k_prep_atomT(const float* __restrict__ atom,
                             bf16_t* __restrict__ atomT) {
  size_t i = (size_t)blockIdx.x * blockDim.x + threadIdx.x;
  if (i >= (size_t)B_ * D_ * N_) return;
  size_t b = i / ((size_t)D_ * N_);
  size_t r = i % ((size_t)D_ * N_);
  size_t d = r / N_, n = r % N_;
  atomT[i] = (bf16_t)atom[(b * N_ + n) * D_ + d];
}

__global__ void k_prep_w(const float* __restrict__ fc_w,
                         const float* __restrict__ kernels,
                         bf16_t* __restrict__ fcwT, bf16_t* __restrict__ kernT) {
  size_t i = (size_t)blockIdx.x * blockDim.x + threadIdx.x;
  if (i >= (size_t)H_ * D_ * D_) return;
  size_t h = i / (D_ * D_);
  size_t r = i % (D_ * D_);
  size_t o = r / D_, k = r % D_;
  kernT[i] = (bf16_t)kernels[(h * D_ + k) * D_ + o];
  if (h == 0) fcwT[r] = (bf16_t)fc_w[k * D_ + o];
}

// ---------------------------------------------------------------------------
// base[b] = x[b] @ atom[b]   (fp32 out, bf16 WMMA)
// grid: B*4 blocks; block handles 64 rows x 256 cols. 8 waves x 8 tiles.
// ---------------------------------------------------------------------------
__global__ void __launch_bounds__(256) k_base(const bf16_t* __restrict__ xbf,
                                              const bf16_t* __restrict__ atomT,
                                              float* __restrict__ base) {
  const int wg = blockIdx.x;
  const int b  = wg >> 2;
  const int rb = (wg & 3) << 6;
  const int lane = threadIdx.x & 31;
  const int wave = threadIdx.x >> 5;
  const int mt = wave & 3;        // M tile within 64-row block
  const int ng = wave >> 2;       // 128-col group
  const int lm = lane & 15;
  const int hg = lane >> 4;
  const int kh = hg << 4;
  const int arow = rb + (mt << 4) + lm;
  const bf16_t* ap = xbf + ((size_t)b * N_ + arow) * N_ + kh;

  v8f acc[8];
#pragma unroll
  for (int t = 0; t < 8; ++t)
#pragma unroll
    for (int e = 0; e < 8; ++e) acc[t][e] = 0.0f;

  for (int k0 = 0; k0 < N_; k0 += 32) {
    v16bf a = *(const v16bf*)(ap + k0);
#pragma unroll
    for (int t = 0; t < 8; ++t) {
      const int nc = (ng << 7) + (t << 4) + lm;
      v16bf bb = *(const v16bf*)(atomT + ((size_t)b * D_ + nc) * N_ + k0 + kh);
      acc[t] = wmma_bf16(a, bb, acc[t]);
    }
  }
  const int mrow = rb + (mt << 4) + (hg << 3);
#pragma unroll
  for (int t = 0; t < 8; ++t) {
    const int nc = (ng << 7) + (t << 4) + lm;
#pragma unroll
    for (int r = 0; r < 8; ++r)
      base[((size_t)b * N_ + mrow + r) * D_ + nc] = acc[t][r];
  }
}

// ---------------------------------------------------------------------------
// Per (h, b, 64 rows):
//   nf   = base + (eps[h]-1)*diag*atom                  (built into A frags)
//   hmid = leaky_relu(nf @ fc_w + fc_b)                 (bf16 in LDS)
//   feat = hmid @ kernels[h] + biases[h]                -> featT (bf16)
//   a_s/a_n = feat . att vectors                        (shuffle reduction)
// ---------------------------------------------------------------------------
__global__ void __launch_bounds__(256) k_feat(
    const float* __restrict__ atom, const float* __restrict__ base,
    const float* __restrict__ diag, const bf16_t* __restrict__ fcwT,
    const float* __restrict__ fc_b, const bf16_t* __restrict__ kernT,
    const float* __restrict__ biases, const float* __restrict__ att_self,
    const float* __restrict__ att_neigh, const float* __restrict__ eps,
    const float* __restrict__ att_biases, bf16_t* __restrict__ featT,
    float* __restrict__ a_s, float* __restrict__ a_n) {
  __shared__ bf16_t hmid[64 * LROW];
  __shared__ float asacc[2][64];
  __shared__ float anacc[2][64];

  const int wg = blockIdx.x;
  const int h  = wg >> 9;           // / (B_*4)
  const int r2 = wg & 511;
  const int b  = r2 >> 2;
  const int rb = (r2 & 3) << 6;
  const int lane = threadIdx.x & 31;
  const int wave = threadIdx.x >> 5;
  const int mt = wave & 3;
  const int ng = wave >> 2;
  const int lm = lane & 15;
  const int hg = lane >> 4;
  const int kh = hg << 4;
  const int arow = rb + (mt << 4) + lm;
  const float dg = (eps[h] - 1.0f) * diag[b * N_ + arow];
  const float* brow = base + ((size_t)b * N_ + arow) * D_ + kh;
  const float* trow = atom + ((size_t)b * N_ + arow) * D_ + kh;

  v8f acc[8];
#pragma unroll
  for (int t = 0; t < 8; ++t)
#pragma unroll
    for (int e = 0; e < 8; ++e) acc[t][e] = 0.0f;

  // -------- GEMM1: nf @ fc_w --------
  for (int k0 = 0; k0 < D_; k0 += 32) {
    v16bf a;
    const float4* b4 = (const float4*)(brow + k0);
    const float4* t4 = (const float4*)(trow + k0);
#pragma unroll
    for (int q = 0; q < 4; ++q) {
      float4 bv = b4[q], tv = t4[q];
      a[4 * q + 0] = (bf16_t)(bv.x + dg * tv.x);
      a[4 * q + 1] = (bf16_t)(bv.y + dg * tv.y);
      a[4 * q + 2] = (bf16_t)(bv.z + dg * tv.z);
      a[4 * q + 3] = (bf16_t)(bv.w + dg * tv.w);
    }
#pragma unroll
    for (int t = 0; t < 8; ++t) {
      const int oc = (ng << 7) + (t << 4) + lm;
      v16bf bb = *(const v16bf*)(fcwT + (size_t)oc * D_ + k0 + kh);
      acc[t] = wmma_bf16(a, bb, acc[t]);
    }
  }
  const int mlocal = (mt << 4) + (hg << 3);
#pragma unroll
  for (int t = 0; t < 8; ++t) {
    const int oc = (ng << 7) + (t << 4) + lm;
    const float fb = fc_b[oc];
#pragma unroll
    for (int r = 0; r < 8; ++r) {
      float v = acc[t][r] + fb;
      v = v > 0.0f ? v : 0.01f * v;          // leaky_relu(0.01)
      hmid[(mlocal + r) * LROW + oc] = (bf16_t)v;
    }
  }
  __syncthreads();

  // -------- GEMM2: hmid @ kernels[h] --------
#pragma unroll
  for (int t = 0; t < 8; ++t)
#pragma unroll
    for (int e = 0; e < 8; ++e) acc[t][e] = 0.0f;

  for (int k0 = 0; k0 < D_; k0 += 32) {
    v16bf a = *(const v16bf*)(&hmid[((mt << 4) + lm) * LROW + k0 + kh]);
#pragma unroll
    for (int t = 0; t < 8; ++t) {
      const int oc = (ng << 7) + (t << 4) + lm;
      v16bf bb = *(const v16bf*)(kernT + ((size_t)h * D_ + oc) * D_ + k0 + kh);
      acc[t] = wmma_bf16(a, bb, acc[t]);
    }
  }

  float rps[8], rpn[8];
#pragma unroll
  for (int r = 0; r < 8; ++r) { rps[r] = 0.0f; rpn[r] = 0.0f; }
#pragma unroll
  for (int t = 0; t < 8; ++t) {
    const int oc = (ng << 7) + (t << 4) + lm;
    const float bv   = biases[h * D_ + oc];
    const float as_o = att_self[h * D_ + oc];
    const float an_o = att_neigh[h * D_ + oc];
#pragma unroll
    for (int r = 0; r < 8; ++r) {
      float f = acc[t][r] + bv;
      featT[(((size_t)h * B_ + b) * D_ + oc) * N_ + rb + mlocal + r] = (bf16_t)f;
      rps[r] += f * as_o;
      rpn[r] += f * an_o;
    }
  }
  // deterministic width-16 reduction, one writer per (ng, row)
#pragma unroll
  for (int r = 0; r < 8; ++r) {
    float s = rps[r], n = rpn[r];
#pragma unroll
    for (int d = 8; d >= 1; d >>= 1) {
      s += __shfl_xor(s, d, 16);
      n += __shfl_xor(n, d, 16);
    }
    if (lm == 0) {
      asacc[ng][mlocal + r] = s;
      anacc[ng][mlocal + r] = n;
    }
  }
  __syncthreads();
  if (threadIdx.x < 64) {
    const size_t o = ((size_t)h * B_ + b) * N_ + rb + threadIdx.x;
    a_s[o] = asacc[0][threadIdx.x] + asacc[1][threadIdx.x] + att_biases[h];
    a_n[o] = anacc[0][threadIdx.x] + anacc[1][threadIdx.x];
  }
}

// ---------------------------------------------------------------------------
// Per (h, b, 64 rows): softmax(elu(a_s+a_n)+mask) then node = att @ feat.
// Softmax fp32; unnormalized exp -> bf16 LDS; row scale applied post-GEMM.
// ---------------------------------------------------------------------------
__global__ void __launch_bounds__(256) k_node(
    const float* __restrict__ x, const float* __restrict__ a_s,
    const float* __restrict__ a_n, const bf16_t* __restrict__ featT,
    float* __restrict__ out_node, float* __restrict__ out_att) {
  __shared__ bf16_t patt[64 * LROW];
  __shared__ float ansh[N_];
  __shared__ float rscale[64];

  const int wg = blockIdx.x;
  const int h  = wg >> 9;
  const int r2 = wg & 511;
  const int b  = r2 >> 2;
  const int rb = (r2 & 3) << 6;
  const int tid = threadIdx.x;

  ansh[tid] = a_n[((size_t)h * B_ + b) * N_ + tid];
  __syncthreads();

  if (tid < 64) {
    const int i = rb + tid;
    const float asi = a_s[((size_t)h * B_ + b) * N_ + i];
    const float* xrow = x + ((size_t)b * N_ + i) * N_;
    float mx = -3.0e38f;
    for (int j = 0; j < N_; ++j) {
      float e  = asi + ansh[j];
      float el = e > 0.0f ? e : (expf(e) - 1.0f);
      float v  = el + (xrow[j] > 0.5f ? 0.0f : NEG_BIG_F);
      mx = fmaxf(mx, v);
    }
    float sum = 0.0f;
    for (int j = 0; j < N_; ++j) {
      float e  = asi + ansh[j];
      float el = e > 0.0f ? e : (expf(e) - 1.0f);
      float v  = el + (xrow[j] > 0.5f ? 0.0f : NEG_BIG_F);
      float p  = expf(v - mx);
      sum += p;
      patt[tid * LROW + j] = (bf16_t)p;     // unnormalized; scaled post-GEMM
    }
    const float inv = 1.0f / sum;
    rscale[tid] = inv;
    if (h == H_ - 1) {                       // exact fp32 att[-1] output
      for (int j = 0; j < N_; ++j) {
        float e  = asi + ansh[j];
        float el = e > 0.0f ? e : (expf(e) - 1.0f);
        float v  = el + (xrow[j] > 0.5f ? 0.0f : NEG_BIG_F);
        out_att[((size_t)b * N_ + i) * N_ + j] = expf(v - mx) * inv;
      }
    }
  }
  __syncthreads();

  const int lane = tid & 31;
  const int wave = tid >> 5;
  const int mt = wave & 3;
  const int ng = wave >> 2;
  const int lm = lane & 15;
  const int hg = lane >> 4;
  const int kh = hg << 4;

  v8f acc[8];
#pragma unroll
  for (int t = 0; t < 8; ++t)
#pragma unroll
    for (int e = 0; e < 8; ++e) acc[t][e] = 0.0f;

  for (int k0 = 0; k0 < N_; k0 += 32) {
    v16bf a = *(const v16bf*)(&patt[((mt << 4) + lm) * LROW + k0 + kh]);
#pragma unroll
    for (int t = 0; t < 8; ++t) {
      const int oc = (ng << 7) + (t << 4) + lm;
      v16bf bb =
          *(const v16bf*)(featT + (((size_t)h * B_ + b) * D_ + oc) * N_ + k0 + kh);
      acc[t] = wmma_bf16(a, bb, acc[t]);
    }
  }
  const int mlocal = (mt << 4) + (hg << 3);
#pragma unroll
  for (int t = 0; t < 8; ++t) {
    const int oc = (ng << 7) + (t << 4) + lm;
#pragma unroll
    for (int r = 0; r < 8; ++r) {
      const int i = rb + mlocal + r;
      out_node[(((size_t)b * N_ + i) * H_ + h) * D_ + oc] =
          acc[t][r] * rscale[mlocal + r];
    }
  }
}

// ---------------------------------------------------------------------------
extern "C" void kernel_launch(void* const* d_in, const int* in_sizes, int n_in,
                              void* d_out, int out_size, void* d_ws,
                              size_t ws_size, hipStream_t stream) {
  const float* atom   = (const float*)d_in[0];
  const float* x      = (const float*)d_in[1];
  const float* fc_w   = (const float*)d_in[2];
  const float* fc_b   = (const float*)d_in[3];
  const float* kern   = (const float*)d_in[4];
  const float* att_s  = (const float*)d_in[5];
  const float* att_n  = (const float*)d_in[6];
  const float* eps    = (const float*)d_in[7];
  const float* biases = (const float*)d_in[8];
  const float* att_b  = (const float*)d_in[9];

  float* out_node = (float*)d_out;                          // [B,N,H*D]
  float* out_att  = out_node + (size_t)B_ * N_ * H_ * D_;   // [B,N,N]

  // Workspace layout (~136 MB total; every slice 256B-aligned)
  char* w = (char*)d_ws;
  bf16_t* xbf   = (bf16_t*)w; w += (size_t)B_ * N_ * N_ * sizeof(bf16_t);
  bf16_t* atomT = (bf16_t*)w; w += (size_t)B_ * D_ * N_ * sizeof(bf16_t);
  bf16_t* fcwT  = (bf16_t*)w; w += (size_t)D_ * D_ * sizeof(bf16_t);
  bf16_t* kernT = (bf16_t*)w; w += (size_t)H_ * D_ * D_ * sizeof(bf16_t);
  bf16_t* featT = (bf16_t*)w; w += (size_t)H_ * B_ * D_ * N_ * sizeof(bf16_t);
  float*  diag  = (float*)w;  w += (size_t)B_ * N_ * sizeof(float);
  float*  base  = (float*)w;  w += (size_t)B_ * N_ * D_ * sizeof(float);
  float*  a_sw  = (float*)w;  w += (size_t)H_ * B_ * N_ * sizeof(float);
  float*  a_nw  = (float*)w;  w += (size_t)H_ * B_ * N_ * sizeof(float);

  dim3 blk(256);
  k_prep_xbf  <<<(B_ * N_ * N_) / 256, blk, 0, stream>>>(x, xbf, diag);
  k_prep_atomT<<<(B_ * D_ * N_) / 256, blk, 0, stream>>>(atom, atomT);
  k_prep_w    <<<(H_ * D_ * D_) / 256, blk, 0, stream>>>(fc_w, kern, fcwT, kernT);
  k_base      <<<B_ * 4, blk, 0, stream>>>(xbf, atomT, base);
  k_feat      <<<H_ * B_ * 4, blk, 0, stream>>>(atom, base, diag, fcwT, fc_b,
                                                kernT, biases, att_s, att_n,
                                                eps, att_b, featT, a_sw, a_nw);
  k_node      <<<H_ * B_ * 4, blk, 0, stream>>>(x, a_sw, a_nw, featT, out_node,
                                                out_att);
}
